// VLP_LSTM_LB_v3_32908039422225
// MI455X (gfx1250) — matrix operand order repaired and checked by reading, not compiled, not run on hardware
//
#include <hip/hip_runtime.h>
#include <hip/hip_bf16.h>
#include <math.h>

typedef __attribute__((ext_vector_type(16))) _Float16 v16h;
typedef __attribute__((ext_vector_type(8)))  _Float16 v8h;
typedef __attribute__((ext_vector_type(8)))  float    v8f;
typedef __attribute__((ext_vector_type(4)))  unsigned int u32x4;
typedef __attribute__((ext_vector_type(8)))  int i32x8;
typedef __attribute__((ext_vector_type(4)))  int i32x4;

#define DEV __device__ __forceinline__

// ---------------- problem constants ----------------
constexpr int B_SZ = 512, T = 512, RSSN = 12, H = 128, NG = 512; // NG = 4*H gates
constexpr int ROWS = 64;               // batch rows per main block
constexpr int NBLK = B_SZ / ROWS;      // 8 persistent blocks

// ---------------- workspace layout ----------------
// f16 region (element offsets into (_Float16*)d_ws)
constexpr size_t W0_OFF  = 0;                       // 8Kt x 32Nt frags = 131072 halves
constexpr size_t W1_OFF  = W0_OFF + 131072;         // 131072
constexpr size_t KKH_OFF = W1_OFF + 131072;         // 12 frags   = 6144
constexpr size_t FC1_OFF = KKH_OFF + 6144;          // 16 frags   = 8192
constexpr size_t H0_OFF  = FC1_OFF + 8192;          // 512*128    = 65536
constexpr size_t H1_OFF  = H0_OFF + 65536;          // 65536
constexpr size_t FP_OFF  = H1_OFF + 65536;          // 512*512*96 = 25165824
constexpr size_t F16_ELEMS = FP_OFF + (size_t)512 * 512 * 96;
constexpr size_t F16_BYTES = F16_ELEMS * 2;
// f32 region (element offsets into (float*)((char*)d_ws + F16_BYTES))
constexpr size_t BIAS0_F = 0;                       // 512
constexpr size_t BIAS1_F = 512;                     // 512
constexpr size_t KK_F    = 1024;                    // 36*137
constexpr size_t KB_F    = KK_F + 36 * 137;         // 36
constexpr size_t LEDF_F  = KB_F + 36;               // 36*8
constexpr size_t C0_F    = LEDF_F + 288;            // 512*128
constexpr size_t C1_F    = C0_F + 65536;            // 512*128

// ---------------- TDM availability ----------------
#if __has_builtin(__builtin_amdgcn_tensor_load_to_lds)
#define ENABLE_TDM 1
#else
#define ENABLE_TDM 0
#endif

DEV float sigm(float x) { return 1.0f / (1.0f + expf(-x)); }

// WMMA wrapper: D = A(16x32 f16) * B(32x16 f16) + C(f32)
DEV v8f wmma_f16(v16h a, v16h b, v8f c) {
  return __builtin_amdgcn_wmma_f32_16x16x32_f16(false, a, false, b, (short)0, c,
                                                false, false);
}

// A fragment (16x32, f16) from an LDS tile with row-stride ldk (halves).
// Layout (ISA 7.12.2): lane<16 -> K 0..7 & 16..23, lane>=16 -> +8.
DEV v16h load_A_frag(const _Float16* tile, int ldk, int kt, int lane) {
  int m = lane & 15;
  int koff = kt * 32 + ((lane & 16) ? 8 : 0);
  const _Float16* p = tile + m * ldk + koff;
  v8h lo = *(const v8h*)p;
  v8h hi = *(const v8h*)(p + 16);
  v16h r;
#pragma unroll
  for (int e = 0; e < 8; ++e) { r[e] = lo[e]; r[e + 8] = hi[e]; }
  return r;
}

// B fragment: pre-packed (k_pack) so each lane's 16 halves are contiguous.
DEV v16h load_B_frag(const _Float16* fragbase, int lane) {
  return *(const v16h*)(fragbase + lane * 16);
}

#if ENABLE_TDM
// 2-D tensor DMA: global (tile_d1 rows x tile_d0 f16, row stride in f16 units) -> LDS
DEV void tdm_load_tile_f16(unsigned lds_byte_off, const void* gsrc,
                           unsigned tile_d0, unsigned tile_d1,
                           unsigned long long row_stride_elems) {
  unsigned long long ga = (unsigned long long)(uintptr_t)gsrc;
  u32x4 g0;
  g0[0] = 1u;                                              // count=1 (user D#)
  g0[1] = lds_byte_off;                                    // lds_addr
  g0[2] = (unsigned)ga;                                    // global_addr[31:0]
  g0[3] = (unsigned)((ga >> 32) & 0x01FFFFFFu) | (2u << 30); // addr[56:32] | type=2
  i32x8 g1;
  g1[0] = 0x00010000;                                      // data_size=1 (2B)
  g1[1] = (int)((tile_d0 & 0xFFFFu) << 16);                // tensor_dim0 lo -> [63:48]
  g1[2] = (int)(((tile_d0 >> 16) & 0xFFFFu) | ((tile_d1 & 0xFFFFu) << 16));
  g1[3] = (int)(((tile_d1 >> 16) & 0xFFFFu) | ((tile_d0 & 0xFFFFu) << 16)); // tile_dim0
  g1[4] = (int)(tile_d1 & 0xFFFFu);                        // tile_dim1 (tile_dim2=0)
  g1[5] = (int)(row_stride_elems & 0xFFFFFFFFull);         // dim0_stride[31:0]
  g1[6] = (int)((row_stride_elems >> 32) & 0xFFFFull);     // dim0_stride[47:32]
  g1[7] = 0;
  i32x4 z4 = {0, 0, 0, 0};
#if defined(__clang_major__) && (__clang_major__ >= 23)
  i32x8 z8 = {0, 0, 0, 0, 0, 0, 0, 0};
  __builtin_amdgcn_tensor_load_to_lds(g0, g1, z4, z4, z8, 0);
#else
  __builtin_amdgcn_tensor_load_to_lds(g0, g1, z4, z4, 0);
#endif
}
#endif

// =====================================================================
// Kernel 1: graph precompute (GCN, keys, folded KK = query_w @ keys^T)
// =====================================================================
__global__ void k_prep_graph(const float* __restrict__ led_pos_freq,
                             const float* __restrict__ led_B,
                             const float* __restrict__ freq_emb,
                             const float* __restrict__ gcn1_w, const float* __restrict__ gcn1_b,
                             const float* __restrict__ gcn2_w, const float* __restrict__ gcn2_b,
                             const float* __restrict__ key_w, const float* __restrict__ key_b,
                             const float* __restrict__ query_w, const float* __restrict__ query_b,
                             float* __restrict__ wsf) {
  __shared__ float pos[36 * 3];
  __shared__ int   freq[36];
  __shared__ float x[36 * 40];
  __shared__ float adjn[36 * 36];
  __shared__ float t1[36 * 32];
  __shared__ float t2[36 * 32];
  __shared__ float ledf[36 * 8];
  __shared__ float keys[36 * 64];
  const int tid = threadIdx.x;
  if (tid < 36) {
    for (int k = 0; k < 3; ++k) pos[tid * 3 + k] = led_pos_freq[tid * 4 + k];
    freq[tid] = (int)led_pos_freq[tid * 4 + 3];
  }
  __syncthreads();
  for (int u = tid; u < 36 * 16; u += 256) {
    int i = u / 16, j = u % 16;
    float pr = 0.f;
    for (int k = 0; k < 3; ++k) pr += pos[i * 3 + k] * led_B[k * 16 + j];
    pr *= 6.28318530717958647692f;
    x[i * 40 + j] = sinf(pr);
    x[i * 40 + 16 + j] = cosf(pr);
  }
  for (int u = tid; u < 36 * 8; u += 256) {
    int i = u / 8, e = u % 8;
    x[i * 40 + 32 + e] = freq_emb[freq[i] * 8 + e];
  }
  for (int u = tid; u < 36 * 36; u += 256) {
    int i = u / 36, j = u % 36;
    float d2 = 0.f;
    for (int k = 0; k < 3; ++k) { float dd = pos[i * 3 + k] - pos[j * 3 + k]; d2 += dd * dd; }
    adjn[u] = (sqrtf(d2) < 2.5f) ? 1.0f : 0.0f;
  }
  __syncthreads();
  if (tid < 36) {
    float s = 0.f;
    for (int j = 0; j < 36; ++j) s += adjn[tid * 36 + j];
    float inv = 1.0f / (s + 1e-6f);
    for (int j = 0; j < 36; ++j) adjn[tid * 36 + j] *= inv;
  }
  __syncthreads();
  for (int u = tid; u < 36 * 32; u += 256) {
    int i = u / 32, j = u % 32;
    float s = gcn1_b[j];
    for (int k = 0; k < 40; ++k) s += x[i * 40 + k] * gcn1_w[k * 32 + j];
    t1[u] = s;
  }
  __syncthreads();
  for (int u = tid; u < 36 * 32; u += 256) {
    int i = u / 32, j = u % 32;
    float s = 0.f;
    for (int m = 0; m < 36; ++m) s += adjn[i * 36 + m] * t1[m * 32 + j];
    t2[u] = fmaxf(s, 0.f);
  }
  __syncthreads();
  for (int u = tid; u < 36 * 8; u += 256) {
    int i = u / 8, j = u % 8;
    float s = gcn2_b[j];
    for (int k = 0; k < 32; ++k) s += t2[i * 32 + k] * gcn2_w[k * 8 + j];
    t1[u] = s;
  }
  __syncthreads();
  for (int u = tid; u < 36 * 8; u += 256) {
    int i = u / 8, j = u % 8;
    float s = 0.f;
    for (int m = 0; m < 36; ++m) s += adjn[i * 36 + m] * t1[m * 8 + j];
    float v = fmaxf(s, 0.f);
    ledf[u] = v;
    wsf[LEDF_F + u] = v;
  }
  __syncthreads();
  for (int u = tid; u < 36 * 64; u += 256) {
    int i = u / 64, j = u % 64;
    float s = key_b[j];
    for (int k = 0; k < 8; ++k) s += ledf[i * 8 + k] * key_w[k * 64 + j];
    for (int k = 0; k < 3; ++k) s += pos[i * 3 + k] * key_w[(8 + k) * 64 + j];
    keys[u] = s;
  }
  __syncthreads();
  for (int u = tid; u < 36 * 137; u += 256) {
    int j = u / 137, i2 = u % 137;
    float s = 0.f;
    for (int d = 0; d < 64; ++d) s += query_w[i2 * 64 + d] * keys[j * 64 + d];
    wsf[KK_F + u] = s;
  }
  if (tid < 36) {
    float s = 0.f;
    for (int d = 0; d < 64; ++d) s += query_b[d] * keys[tid * 64 + d];
    wsf[KB_F + tid] = s;
  }
}

// =====================================================================
// Kernel 2: initial LSTM state h0/h1 (f16) and c0/c1 (f32)
// =====================================================================
__global__ void k_init_state(const float* __restrict__ init_pos,
                             const float* __restrict__ w1, const float* __restrict__ b1,
                             const float* __restrict__ w2, const float* __restrict__ b2,
                             _Float16* __restrict__ wsh, float* __restrict__ wsf) {
  __shared__ float hid[64];
  const int b = blockIdx.x, tid = threadIdx.x;
  float s = b1[tid];
  for (int k = 0; k < 3; ++k) s += init_pos[b * 3 + k] * w1[k * 64 + tid];
  hid[tid] = fmaxf(s, 0.f);
  __syncthreads();
  for (int q = 0; q < 8; ++q) {
    int n = tid * 8 + q;
    float v = b2[n];
    for (int j = 0; j < 64; ++j) v += hid[j] * w2[j * 512 + n];
    int seg = n >> 7, h = n & 127;
    if (seg == 0)      wsh[H0_OFF + (size_t)b * H + h] = (_Float16)v;
    else if (seg == 1) wsh[H1_OFF + (size_t)b * H + h] = (_Float16)v;
    else if (seg == 2) wsf[C0_F + (size_t)b * H + h] = v;
    else               wsf[C1_F + (size_t)b * H + h] = v;
  }
}

// =====================================================================
// Kernel 3: conv features fp[b,t,12,8] (f16), one thread per (b,t)
// =====================================================================
__global__ void k_conv_fp(const float* __restrict__ rss_seq,
                          const float* __restrict__ w1, const float* __restrict__ b1,
                          const float* __restrict__ w2, const float* __restrict__ b2,
                          _Float16* __restrict__ wsh) {
  __shared__ float sw1[48], sb1[16], sw2[384], sb2[8];
  const int tid = threadIdx.x;
  if (tid < 48) sw1[tid] = w1[tid];
  if (tid < 16) sb1[tid] = b1[tid];
  if (tid < 8)  sb2[tid] = b2[tid];
  for (int u = tid; u < 384; u += 256) sw2[u] = w2[u];
  __syncthreads();
  const size_t idx = (size_t)blockIdx.x * 256 + tid;  // b*T + t
  float in[12];
#pragma unroll
  for (int r = 0; r < 12; ++r) in[r] = rss_seq[idx * 12 + r];
  auto getin = [&](int r) -> float { return (r >= 0 && r < 12) ? in[r] : 0.f; };
  float colm[16], col0[16], colp[16];
#pragma unroll
  for (int o = 0; o < 16; ++o) colm[o] = 0.f;
#pragma unroll
  for (int o = 0; o < 16; ++o)
    col0[o] = fmaxf(sb1[o] + sw1[o * 3 + 1] * in[0] + sw1[o * 3 + 2] * in[1], 0.f);
  _Float16* dst = wsh + FP_OFF + idx * 96;
  for (int r = 0; r < 12; ++r) {
    if (r + 1 < 12) {
#pragma unroll
      for (int o = 0; o < 16; ++o)
        colp[o] = fmaxf(sb1[o] + sw1[o * 3 + 0] * getin(r) + sw1[o * 3 + 1] * in[r + 1] +
                            sw1[o * 3 + 2] * getin(r + 2), 0.f);
    } else {
#pragma unroll
      for (int o = 0; o < 16; ++o) colp[o] = 0.f;
    }
#pragma unroll
    for (int o2 = 0; o2 < 8; ++o2) {
      float a = sb2[o2];
#pragma unroll
      for (int i = 0; i < 16; ++i)
        a += sw2[(o2 * 16 + i) * 3 + 0] * colm[i] + sw2[(o2 * 16 + i) * 3 + 1] * col0[i] +
             sw2[(o2 * 16 + i) * 3 + 2] * colp[i];
      dst[r * 8 + o2] = (_Float16)fmaxf(a, 0.f);
    }
#pragma unroll
    for (int o = 0; o < 16; ++o) { colm[o] = col0[o]; col0[o] = colp[o]; }
  }
}

// =====================================================================
// Kernel 4: pack all B-operands into CDNA5 WMMA fragment layout (f16)
// frag order: nt*KT + kt; within frag: lane*16 + e; k = 32kt+16*(e>=8)+8*(lane>=16)+(e&7)
// =====================================================================
constexpr int NW0 = 131072, NW1 = 131072, NKKH = 6144, NFC1 = 8192;
constexpr int NPACK = NW0 + NW1 + NKKH + NFC1;
__global__ void k_pack(const float* __restrict__ Wih0, const float* __restrict__ Whh0,
                       const float* __restrict__ Wih1, const float* __restrict__ Whh1,
                       const float* __restrict__ fc1_w,
                       const float* __restrict__ bih0, const float* __restrict__ bhh0,
                       const float* __restrict__ bih1, const float* __restrict__ bhh1,
                       _Float16* __restrict__ wsh, float* __restrict__ wsf) {
  const int idx = blockIdx.x * 256 + threadIdx.x;
  if (idx < NPACK) {
    const float* KK = wsf + KK_F;
    _Float16* dst; int em, KT, which;
    if (idx < NW0)                   { which = 0; em = idx;                     dst = wsh + W0_OFF;  KT = 8; }
    else if (idx < NW0 + NW1)        { which = 1; em = idx - NW0;               dst = wsh + W1_OFF;  KT = 8; }
    else if (idx < NW0 + NW1 + NKKH) { which = 2; em = idx - NW0 - NW1;         dst = wsh + KKH_OFF; KT = 4; }
    else                             { which = 3; em = idx - NW0 - NW1 - NKKH;  dst = wsh + FC1_OFF; KT = 4; }
    int frag = em >> 9;
    int lid = (em >> 4) & 31;
    int e = em & 15;
    int nt = frag / KT, kt = frag % KT;
    int k = kt * 32 + ((e >> 3) & 1) * 16 + ((lid >> 4) & 1) * 8 + (e & 7);
    int n = nt * 16 + (lid & 15);
    float v = 0.f;
    if (which == 0) {           // [236(+pad)x512]: fused108 | h128
      if (k < 108)      v = Wih0[n * 108 + k];
      else if (k < 236) v = Whh0[n * 128 + (k - 108)];
    } else if (which == 1) {    // [256x512]: h0n 128 | h1_prev 128
      if (k < 128) v = Wih1[n * 128 + k];
      else         v = Whh1[n * 128 + (k - 128)];
    } else if (which == 2) {    // KKh [128 x 36(+pad)]
      if (n < 36 && k < 128) v = KK[n * 137 + 9 + k];
    } else {                    // fc1 [128 x 64]
      if (k < 128 && n < 64) v = fc1_w[k * 64 + n];
    }
    dst[em] = (_Float16)v;
  } else if (idx < NPACK + 1024) {
    int j = idx - NPACK;
    if (j < 512) wsf[BIAS0_F + j] = bih0[j] + bhh0[j];
    else         wsf[BIAS1_F + (j - 512)] = bih1[j - 512] + bhh1[j - 512];
  }
}

// Gate GEMM for one LSTM layer: [64 x 256] (LDS f16) @ [256 x 512] (packed) -> sG f32
DEV void gate_gemm(const _Float16* sA, float* sGo, const _Float16* W,
                   const float* bias, int lane, int wave) {
  const v8f zero = {0, 0, 0, 0, 0, 0, 0, 0};
  for (int nn = 0; nn < 4; ++nn) {
    const int nt = wave * 4 + nn;
    const _Float16* bb = W + (size_t)nt * (8 * 512);
    v8f acc[4];
#pragma unroll
    for (int mt = 0; mt < 4; ++mt) acc[mt] = zero;
    for (int kt = 0; kt < 8; ++kt) {
      v16h bf = load_B_frag(bb + kt * 512, lane);
#pragma unroll
      for (int mt = 0; mt < 4; ++mt) {
        v16h af = load_A_frag(sA + mt * 16 * 256, 256, kt, lane);
        acc[mt] = wmma_f16(af, bf, acc[mt]);
      }
    }
    const int n = nt * 16 + (lane & 15);
    const float bv = bias[n];
    const int ro = (lane & 16) ? 8 : 0;
#pragma unroll
    for (int mt = 0; mt < 4; ++mt)
#pragma unroll
      for (int r = 0; r < 8; ++r)
        sGo[(mt * 16 + ro + r) * NG + n] = acc[mt][r] + bv;
  }
}

// =====================================================================
// Main persistent kernel: 8 blocks x 64 batch rows, loop over T=512
// =====================================================================
__global__ void __launch_bounds__(256, 1)
k_main(const float* __restrict__ rss_seq,
       const _Float16* __restrict__ wsh, const float* __restrict__ wsf,
       const float* __restrict__ ln_g, const float* __restrict__ ln_b,
       const float* __restrict__ fc1_b, const float* __restrict__ fc2_w,
       const float* __restrict__ fc2_b, float* __restrict__ out) {
  extern __shared__ char smem[];
  _Float16* sA0 = (_Float16*)(smem + 0);        // 64x256 f16 : fused108|h0|pad
  _Float16* sA1 = (_Float16*)(smem + 32768);    // 64x256 f16 : h0n|h1_prev
  float*    sG  = (float*)(smem + 65536);       // 64x512 f32 gates
  float*    sSc = (float*)(smem + 196608);      // 64x48 f32 h-scores
  _Float16* sY  = (_Float16*)(smem + 208896);   // 64x128 f16 LN output
  _Float16* sZ  = (_Float16*)(smem + 225280);   // 64x64 f16 fc1 output
  _Float16* sFp = (_Float16*)(smem + 233472);   // 64x96 f16 conv features
  float*    sRss = (float*)(smem + 245760);     // 64x12 f32
  constexpr unsigned FP_LDS_OFF = 233472;

  const int tid = threadIdx.x;
  const int lane = tid & 31;
  const int wave = tid >> 5;
  const int b0 = blockIdx.x * ROWS;

  const _Float16* W0 = wsh + W0_OFF;
  const _Float16* W1 = wsh + W1_OFF;
  const _Float16* KKH = wsh + KKH_OFF;
  const _Float16* FC1 = wsh + FC1_OFF;
  const _Float16* FPT = wsh + FP_OFF;
  const float* bias0 = wsf + BIAS0_F;
  const float* bias1 = wsf + BIAS1_F;
  const float* KK = wsf + KK_F;
  const float* kb = wsf + KB_F;
  const float* ledf = wsf + LEDF_F;

  // persistent per-thread cell state: row = tid/4, h in [hb, hb+32)
  const int srow = tid >> 2;
  const int hb = (tid & 3) * 32;
  float c0v[32], c1v[32];
#pragma unroll
  for (int u = 0; u < 32; ++u) {
    c0v[u] = wsf[C0_F + (size_t)(b0 + srow) * H + hb + u];
    c1v[u] = wsf[C1_F + (size_t)(b0 + srow) * H + hb + u];
  }
  for (int u = tid; u < ROWS * H; u += 256) {
    int r = u >> 7, h = u & 127;
    sA0[r * 256 + 108 + h] = wsh[H0_OFF + (size_t)(b0 + r) * H + h];
    sA1[r * 256 + 128 + h] = wsh[H1_OFF + (size_t)(b0 + r) * H + h];
  }
  for (int u = tid; u < ROWS * 20; u += 256) {
    int r = u / 20, j = u % 20;
    sA0[r * 256 + 236 + j] = (_Float16)0.0f;
  }
  __syncthreads();

  const v8f zero = {0, 0, 0, 0, 0, 0, 0, 0};

  for (int t = 0; t < T; ++t) {
    // ---- P0: async tensor-DMA of fp_t tile; rss_t loads + prefetch ----
#if ENABLE_TDM
    if (wave == 0)
      tdm_load_tile_f16(FP_LDS_OFF, (const void*)(FPT + ((size_t)b0 * T + t) * 96),
                        96u, (unsigned)ROWS, (unsigned long long)T * 96);
#else
    for (int u = tid; u < ROWS * 96; u += 256) {
      int r = u / 96, e = u % 96;
      sFp[u] = FPT[((size_t)(b0 + r) * T + t) * 96 + e];
    }
#endif
    for (int u = tid; u < ROWS * RSSN; u += 256) {
      int r = u / RSSN, q = u % RSSN;
      sRss[u] = rss_seq[((size_t)(b0 + r) * T + t) * RSSN + q];
      if (t + 1 < T)
        __builtin_prefetch(&rss_seq[((size_t)(b0 + r) * T + t + 1) * RSSN + q], 0, 1);
    }
    // ---- P1: h-scores = h1_prev @ KKh^T  (A = sA1 kt 4..7) ----
    for (int tile = wave; tile < 12; tile += 8) {
      int mt = tile / 3, nt = tile % 3;
      v8f acc = zero;
#pragma unroll
      for (int kt = 0; kt < 4; ++kt) {
        v16h bf = load_B_frag(KKH + (size_t)(nt * 4 + kt) * 512, lane);
        v16h af = load_A_frag(sA1 + mt * 16 * 256, 256, kt + 4, lane);
        acc = wmma_f16(af, bf, acc);
      }
      int n = nt * 16 + (lane & 15);
      int rowb = mt * 16 + ((lane & 16) ? 8 : 0);
#pragma unroll
      for (int r = 0; r < 8; ++r) sSc[(rowb + r) * 48 + n] = acc[r];
    }
#if ENABLE_TDM
#if __has_builtin(__builtin_amdgcn_s_wait_tensorcnt)
    if (wave == 0) __builtin_amdgcn_s_wait_tensorcnt(0);
#endif
#endif
    __syncthreads();
    // ---- P2: 3-way masked softmax + fused input -> sA0[0..107] ----
    for (int u = tid; u < ROWS * RSSN; u += 256) {
      int r = u / RSSN, cch = u % RSSN;
      float rs = sRss[u];
      float fpv[8];
#pragma unroll
      for (int e = 0; e < 8; ++e) fpv[e] = (float)sFp[u * 8 + e];
      float sc[3];
#pragma unroll
      for (int jj = 0; jj < 3; ++jj) {
        int j = cch + 12 * jj;
        float s = sSc[r * 48 + j] + kb[j] + rs * KK[j * 137 + 0];
#pragma unroll
        for (int e = 0; e < 8; ++e) s += fpv[e] * KK[j * 137 + 1 + e];
        sc[jj] = s;
      }
      float mx = fmaxf(sc[0], fmaxf(sc[1], sc[2]));
      float e0 = expf(sc[0] - mx), e1 = expf(sc[1] - mx), e2 = expf(sc[2] - mx);
      float inv = 1.0f / (e0 + e1 + e2);
      _Float16* dst = sA0 + r * 256 + cch * 9;
      dst[0] = (_Float16)rs;
#pragma unroll
      for (int e = 0; e < 8; ++e) {
        float v = e0 * ledf[cch * 8 + e] + e1 * ledf[(cch + 12) * 8 + e] +
                  e2 * ledf[(cch + 24) * 8 + e];
        dst[1 + e] = (_Float16)(v * inv);
      }
    }
    __syncthreads();
    // ---- P3: LSTM layer 0 gates ----
    gate_gemm(sA0, sG, W0, bias0, lane, wave);
    __syncthreads();
    // ---- P4: layer-0 cell update ----
    {
      const float* g = sG + srow * NG;
#pragma unroll
      for (int u = 0; u < 32; ++u) {
        int h = hb + u;
        float gi = g[h], gf = g[128 + h], gg = g[256 + h], go = g[384 + h];
        float cn = sigm(gf) * c0v[u] + sigm(gi) * tanhf(gg);
        c0v[u] = cn;
        float hn = sigm(go) * tanhf(cn);
        sA1[srow * 256 + h] = (_Float16)hn;        // input to layer 1
        sA0[srow * 256 + 108 + h] = (_Float16)hn;  // h0_prev for next step
      }
    }
    __syncthreads();
    // ---- P5: LSTM layer 1 gates ----
    gate_gemm(sA1, sG, W1, bias1, lane, wave);
    __syncthreads();
    // ---- P6: layer-1 cell update + LayerNorm (4-lane row reduce) ----
    {
      const float* g = sG + srow * NG;
      float hv[32];
      float sum = 0.f, sq = 0.f;
#pragma unroll
      for (int u = 0; u < 32; ++u) {
        int h = hb + u;
        float gi = g[h], gf = g[128 + h], gg = g[256 + h], go = g[384 + h];
        float cn = sigm(gf) * c1v[u] + sigm(gi) * tanhf(gg);
        c1v[u] = cn;
        float hn = sigm(go) * tanhf(cn);
        hv[u] = hn; sum += hn; sq += hn * hn;
        sA1[srow * 256 + 128 + h] = (_Float16)hn;  // h1_prev for next step
      }
      sum += __shfl_xor(sum, 1); sum += __shfl_xor(sum, 2);
      sq  += __shfl_xor(sq, 1);  sq  += __shfl_xor(sq, 2);
      float mu = sum * (1.0f / 128.0f);
      float var = sq * (1.0f / 128.0f) - mu * mu;
      float rstd = rsqrtf(var + 1e-5f);
#pragma unroll
      for (int u = 0; u < 32; ++u) {
        int h = hb + u;
        float y = (hv[u] - mu) * rstd * ln_g[h] + ln_b[h];
        sY[srow * H + h] = (_Float16)y;
      }
    }
    __syncthreads();
    // ---- P7: fc1 (relu) via WMMA ----
    for (int tile = wave; tile < 16; tile += 8) {
      int mt = tile >> 2, nt = tile & 3;
      v8f acc = zero;
#pragma unroll
      for (int kt = 0; kt < 4; ++kt) {
        v16h bf = load_B_frag(FC1 + (size_t)(nt * 4 + kt) * 512, lane);
        v16h af = load_A_frag(sY + mt * 16 * H, H, kt, lane);
        acc = wmma_f16(af, bf, acc);
      }
      int n = nt * 16 + (lane & 15);
      float bv = fc1_b[n];
      int ro = (lane & 16) ? 8 : 0;
#pragma unroll
      for (int r = 0; r < 8; ++r)
        sZ[(mt * 16 + ro + r) * 64 + n] = (_Float16)fmaxf(acc[r] + bv, 0.f);
    }
    __syncthreads();
    // ---- P8: fc2 -> out[b, t, 0..2] ----
    for (int u = tid; u < ROWS * 3; u += 256) {
      int r = u / 3, o = u % 3;
      float acc = fc2_b[o];
      const _Float16* z = sZ + r * 64;
      for (int k2 = 0; k2 < 64; ++k2) acc += (float)z[k2] * fc2_w[k2 * 3 + o];
      out[((size_t)(b0 + r) * T + t) * 3 + o] = acc;
    }
    __syncthreads();
  }
}

// =====================================================================
// d_in order: rss_seq, init_pos, params{freq_emb, gcn1_w, gcn1_b, gcn2_w,
//  gcn2_b, conv1_w, conv1_b, conv2_w, conv2_b, init_w1, init_b1, init_w2,
//  init_b2, query_w, query_b, key_w, key_b, Wih0, Whh0, bih0, bhh0, Wih1,
//  Whh1, bih1, bhh1, ln_g, ln_b, fc1_w, fc1_b, fc2_w, fc2_b},
//  led_pos_freq, led_B
// =====================================================================
extern "C" void kernel_launch(void* const* d_in, const int* in_sizes, int n_in,
                              void* d_out, int out_size, void* d_ws, size_t ws_size,
                              hipStream_t stream) {
  (void)in_sizes; (void)n_in; (void)out_size; (void)ws_size;
  const float* rss_seq  = (const float*)d_in[0];
  const float* init_pos = (const float*)d_in[1];
  const float* freq_emb = (const float*)d_in[2];
  const float* gcn1_w = (const float*)d_in[3];
  const float* gcn1_b = (const float*)d_in[4];
  const float* gcn2_w = (const float*)d_in[5];
  const float* gcn2_b = (const float*)d_in[6];
  const float* conv1_w = (const float*)d_in[7];
  const float* conv1_b = (const float*)d_in[8];
  const float* conv2_w = (const float*)d_in[9];
  const float* conv2_b = (const float*)d_in[10];
  const float* init_w1 = (const float*)d_in[11];
  const float* init_b1 = (const float*)d_in[12];
  const float* init_w2 = (const float*)d_in[13];
  const float* init_b2 = (const float*)d_in[14];
  const float* query_w = (const float*)d_in[15];
  const float* query_b = (const float*)d_in[16];
  const float* key_w = (const float*)d_in[17];
  const float* key_b = (const float*)d_in[18];
  const float* Wih0 = (const float*)d_in[19];
  const float* Whh0 = (const float*)d_in[20];
  const float* bih0 = (const float*)d_in[21];
  const float* bhh0 = (const float*)d_in[22];
  const float* Wih1 = (const float*)d_in[23];
  const float* Whh1 = (const float*)d_in[24];
  const float* bih1 = (const float*)d_in[25];
  const float* bhh1 = (const float*)d_in[26];
  const float* ln_g = (const float*)d_in[27];
  const float* ln_b = (const float*)d_in[28];
  const float* fc1_w = (const float*)d_in[29];
  const float* fc1_b = (const float*)d_in[30];
  const float* fc2_w = (const float*)d_in[31];
  const float* fc2_b = (const float*)d_in[32];
  const float* led_pos_freq = (const float*)d_in[33];
  const float* led_B = (const float*)d_in[34];

  _Float16* wsh = (_Float16*)d_ws;
  float* wsf = (float*)((char*)d_ws + F16_BYTES);

  k_prep_graph<<<1, 256, 0, stream>>>(led_pos_freq, led_B, freq_emb, gcn1_w, gcn1_b,
                                      gcn2_w, gcn2_b, key_w, key_b, query_w, query_b, wsf);
  k_init_state<<<512, 64, 0, stream>>>(init_pos, init_w1, init_b1, init_w2, init_b2, wsh, wsf);
  k_conv_fp<<<1024, 256, 0, stream>>>(rss_seq, conv1_w, conv1_b, conv2_w, conv2_b, wsh);
  k_pack<<<(NPACK + 1024 + 255) / 256, 256, 0, stream>>>(Wih0, Whh0, Wih1, Whh1, fc1_w,
                                                         bih0, bhh0, bih1, bhh1, wsh, wsf);
  constexpr int SMEM_MAIN = 248832;  // 243 KB dynamic LDS (< 320 KB/WGP)
  (void)hipFuncSetAttribute(reinterpret_cast<const void*>(k_main),
                            hipFuncAttributeMaxDynamicSharedMemorySize, SMEM_MAIN);
  k_main<<<NBLK, 256, SMEM_MAIN, stream>>>(rss_seq, wsh, wsf, ln_g, ln_b, fc1_b,
                                           fc2_w, fc2_b, (float*)d_out);
}
